// Encoder_50998441673231
// MI455X (gfx1250) — compile-verified, matrix operand order checked by reading, not run on hardware
//
#include <hip/hip_runtime.h>

// ---------------------------------------------------------------------------
// Transformer encoder forward for MI455X (gfx1250, wave32, WMMA).
// B=4 S=1024 D=1024 H=16 HD=64 F=4096 L=6
// ---------------------------------------------------------------------------

#define B_  4
#define S_  1024
#define D_  1024
#define H_  16
#define HD_ 64
#define F_  4096
#define L_  6
#define BS_ (B_ * S_)   // 4096 token rows

typedef __bf16 bf16;
typedef __attribute__((ext_vector_type(16))) __bf16 v16bf;
typedef __attribute__((ext_vector_type(8)))  __bf16 v8bf;
typedef __attribute__((ext_vector_type(8)))  float  v8f;
typedef int v4i_ __attribute__((vector_size(16)));   // payload type of b128 async copy

// ---------------------------------------------------------------------------
// Async global->LDS copy (CDNA5 GLOBAL_LOAD_ASYNC_TO_LDS_B128, ASYNCcnt).
// Builtin signature (from compiler diagnostic):
//   (v4i AS(1)* src_global, v4i AS(3)* dst_lds, imm offset, imm cpol)
// ---------------------------------------------------------------------------
#if defined(__HIP_DEVICE_COMPILE__) && \
    __has_builtin(__builtin_amdgcn_global_load_async_to_lds_b128)
#define USE_ASYNC 1
#else
#define USE_ASYNC 0
#endif

__device__ __forceinline__ void cp128_async(const bf16* g, bf16* l) {
#if USE_ASYNC
  __builtin_amdgcn_global_load_async_to_lds_b128(
      (__attribute__((address_space(1))) v4i_*)g,
      (__attribute__((address_space(3))) v4i_*)l, 0, 0);
#else
  *(uint4*)l = *(const uint4*)g;
#endif
}

template <int N>
__device__ __forceinline__ void wait_async() {
#if USE_ASYNC
#if __has_builtin(__builtin_amdgcn_s_wait_asynccnt)
  __builtin_amdgcn_s_wait_asynccnt(N);
#else
  asm volatile("s_wait_asynccnt %0" ::"i"(N) : "memory");
#endif
#endif
}

// ---------------------------------------------------------------------------
// Fragment helpers.
//
// 16-bit A-matrix 16x32 fragment layout (CDNA5 ISA 7.12.2):
//   per-lane elements: a[0..7]  = A[row, lo+0 .. lo+7]
//                      a[8..15] = A[row, lo+16 .. lo+23],  lo = (lane<16)?0:8
// B fragment mirrors this with "row" = output column (K-contiguous storage),
// so both A tiles and (pre-transposed) B tiles load as 2x ds_load_b128.
// ---------------------------------------------------------------------------
__device__ __forceinline__ v16bf load_frag(const bf16* __restrict__ base,
                                           int row, int stride, int kOff) {
  const int lane = threadIdx.x & 31;
  const int lo = (lane < 16) ? 0 : 8;
  const bf16* p = base + (size_t)row * stride + kOff + lo;
  v8bf r0 = *(const v8bf*)(p);
  v8bf r1 = *(const v8bf*)(p + 16);
  v16bf a;
#pragma unroll
  for (int i = 0; i < 8; ++i) { a[i] = r0[i]; a[i + 8] = r1[i]; }
  return a;
}

__device__ __forceinline__ v8f wmma_bf16(v16bf a, v16bf b, v8f c) {
  // D = A(16x32) * B(32x16) + C, fp32 accumulate
  return __builtin_amdgcn_wmma_f32_16x16x32_bf16(
      false, a, false, b, (short)0, c, false, false);
}

// ---------------------------------------------------------------------------
// Weight transpose + bf16 pack: W[K,N] f32 -> Wt[N,K] bf16 (K contiguous).
// ---------------------------------------------------------------------------
__global__ __launch_bounds__(256) void transpose_w_kernel(
    const float* __restrict__ W, bf16* __restrict__ Wt, int K, int N) {
  size_t i = (size_t)blockIdx.x * 256 + threadIdx.x;
  size_t total = (size_t)K * N;
  if (i < total) {
    int n = (int)(i / K);
    int k = (int)(i % K);
    Wt[i] = (bf16)W[(size_t)k * N + n];
  }
}

// ---------------------------------------------------------------------------
// Embedding + sinusoidal positional encoding. Writes f32 (residual stream)
// and bf16 (GEMM operand) activations.
// ---------------------------------------------------------------------------
__global__ __launch_bounds__(256) void embed_pe_kernel(
    const int* __restrict__ tokens, const float* __restrict__ emb,
    float* __restrict__ x32, bf16* __restrict__ xb) {
  const int t = blockIdx.x;           // token index in [0, B*S)
  const int s = t % S_;               // position
  const int tok = tokens[t];
  const size_t base = (size_t)t * D_;
#pragma unroll
  for (int j = 0; j < 4; ++j) {
    int d = threadIdx.x + j * 256;
    float e = emb[(size_t)tok * D_ + d];
    int i2 = d >> 1;
    float denom = __powf(10000.0f, (2.0f * (float)i2) / (float)D_);
    float ang = (float)s / denom;
    float pe = (d & 1) ? __cosf(ang) : __sinf(ang);
    float v = e + pe;
    x32[base + d] = v;
    xb[base + d] = (bf16)v;
  }
}

// ---------------------------------------------------------------------------
// WMMA GEMM: C[M,N] = A[M,K](bf16) * Bt[N,K]^T(bf16) + bias, optional ReLU.
// Block tile 128x256, BK=32, 8 waves (2x4), wave tile 64x64 => 16 WMMA per
// k-step against 16 ds_load_b128 (1:1). Double-buffered LDS staged with
// async global->LDS copies (6 x b128 per thread per tile, ASYNCcnt-waited).
// ---------------------------------------------------------------------------
__global__ __launch_bounds__(256) void gemm_wmma_kernel(
    const bf16* __restrict__ A, const bf16* __restrict__ Bt,
    const float* __restrict__ bias,
    float* __restrict__ Cf, bf16* __restrict__ Cb,
    int M, int N, int K, int relu) {
  __shared__ __align__(16) bf16 sA[2 * 128 * 32];   // 16 KB
  __shared__ __align__(16) bf16 sB[2 * 256 * 32];   // 32 KB

  const int tid = threadIdx.x;
  const int lane = tid & 31;
  const int wave = tid >> 5;
  const int wm = wave & 1;            // 2 waves along M (64 rows each)
  const int wn = wave >> 1;           // 4 waves along N (64 cols each)
  const int ln16 = lane & 15;
  const int rowAdd = (lane >= 16) ? 8 : 0;
  const int m0 = blockIdx.y * 128;
  const int n0 = blockIdx.x * 256;

  v8f acc[4][4];
  v8f vzero = {};
#pragma unroll
  for (int mi = 0; mi < 4; ++mi)
#pragma unroll
    for (int ni = 0; ni < 4; ++ni) acc[mi][ni] = vzero;

  const int arow = tid >> 1;
  const int aseg = (tid & 1) * 16;    // element offset within BK=32

  auto stage = [&](int t, int buf) {
    // A tile: 128x32 (2 x b128 per thread)
    const bf16* ga = A + (size_t)(m0 + arow) * K + t * 32 + aseg;
    bf16* la = sA + buf * (128 * 32) + arow * 32 + aseg;
    cp128_async(ga, la);
    cp128_async(ga + 8, la + 8);
    // B tile: 256x32 (4 x b128 per thread, one full row)
    const bf16* gb = Bt + (size_t)(n0 + tid) * K + t * 32;
    bf16* lb = sB + buf * (256 * 32) + tid * 32;
    cp128_async(gb, lb);
    cp128_async(gb + 8, lb + 8);
    cp128_async(gb + 16, lb + 16);
    cp128_async(gb + 24, lb + 24);
  };

  const int ntiles = K / 32;
  stage(0, 0);

  for (int t = 0; t < ntiles; ++t) {
    const int cur = t & 1;
    if (t + 1 < ntiles) {
      stage(t + 1, cur ^ 1);
      wait_async<6>();    // <=6 outstanding => tile t fully landed
    } else {
      wait_async<0>();
    }
    __syncthreads();

    const bf16* cA = sA + cur * (128 * 32);
    const bf16* cB = sB + cur * (256 * 32);
    v16bf af[4];
#pragma unroll
    for (int mi = 0; mi < 4; ++mi)
      af[mi] = load_frag(cA, wm * 64 + mi * 16 + ln16, 32, 0);
#pragma unroll
    for (int ni = 0; ni < 4; ++ni) {
      v16bf bfr = load_frag(cB, wn * 64 + ni * 16 + ln16, 32, 0);
#pragma unroll
      for (int mi = 0; mi < 4; ++mi)
        acc[mi][ni] = wmma_bf16(af[mi], bfr, acc[mi][ni]);
    }
    __syncthreads();
  }

  // --- epilogue: bias (+ReLU), write f32/bf16
#pragma unroll
  for (int mi = 0; mi < 4; ++mi) {
#pragma unroll
    for (int ni = 0; ni < 4; ++ni) {
      const int col = n0 + wn * 64 + ni * 16 + ln16;
      const float bv = bias[col];
#pragma unroll
      for (int r = 0; r < 8; ++r) {
        const int row = m0 + wm * 64 + mi * 16 + r + rowAdd;
        float v = acc[mi][ni][r] + bv;
        if (relu) v = fmaxf(v, 0.0f);
        if (Cf) Cf[(size_t)row * N + col] = v;
        if (Cb) Cb[(size_t)row * N + col] = (bf16)v;
      }
    }
  }
}

// ---------------------------------------------------------------------------
// Flash attention: grid (S/128, B*H), 8 waves, each wave owns 16 query rows.
// Q fragments pinned in VGPRs; 32-key K/V tiles staged in LDS (K via async
// copy, V transposed at staging); online softmax with shfl_xor reductions;
// P routed through per-wave LDS. Output bf16 in [B,S,D].
// ---------------------------------------------------------------------------
__global__ __launch_bounds__(256) void attn_kernel(
    const bf16* __restrict__ qkv,     // [B,S,3D], per row: H x (q64|k64|v64)
    const float* __restrict__ mask,   // [B,S,S]
    bf16* __restrict__ vals) {        // [B,S,D]
  __shared__ __align__(16) bf16 sK[32 * 64];   // 32 keys x 64 dims
  __shared__ __align__(16) bf16 sVt[64 * 32];  // 64 dims x 32 keys (transposed)
  __shared__ __align__(16) bf16 sP[8][16 * 32];

  const int tid = threadIdx.x;
  const int lane = tid & 31;
  const int w = tid >> 5;
  const int ln16 = lane & 15;
  const int rowAdd = (lane >= 16) ? 8 : 0;
  const int qchunk = blockIdx.x;
  const int b = blockIdx.y >> 4;      // / H
  const int h = blockIdx.y & 15;

  const int qrow = qchunk * 128 + w * 16 + ln16;
  const bf16* qp = qkv + ((size_t)(b * S_ + qrow)) * (3 * D_) + h * (3 * HD_);
  const v16bf q0 = load_frag(qp, 0, 0, 0);    // dims 0..31
  const v16bf q1 = load_frag(qp, 0, 0, 32);   // dims 32..63

  float mrow[8], lrow[8];
  v8f o[4];
  v8f vzero = {};
#pragma unroll
  for (int r = 0; r < 8; ++r) { mrow[r] = -1e30f; lrow[r] = 0.0f; }
#pragma unroll
  for (int n = 0; n < 4; ++n) o[n] = vzero;

  const int sidx = tid * 8;
  const int skey = sidx >> 6;
  const int sd0 = sidx & 63;

  for (int kt = 0; kt < S_; kt += 32) {
    __syncthreads();
    // --- stage K tile (async) and transposed V tile
    {
      const size_t rbase =
          ((size_t)(b * S_ + kt + skey)) * (3 * D_) + h * (3 * HD_);
      cp128_async(qkv + rbase + HD_ + sd0, sK + skey * 64 + sd0);
      uint4 vv = *(const uint4*)(qkv + rbase + 2 * HD_ + sd0);
      const bf16* ve = (const bf16*)&vv;
#pragma unroll
      for (int j = 0; j < 8; ++j) sVt[(sd0 + j) * 32 + skey] = ve[j];
      if (kt + 32 < S_) {  // prefetch next tile's K/V rows into cache
        __builtin_prefetch(qkv + rbase + (size_t)32 * 3 * D_ + HD_ + sd0, 0, 3);
        __builtin_prefetch(qkv + rbase + (size_t)32 * 3 * D_ + 2 * HD_ + sd0, 0, 3);
      }
    }
    wait_async<0>();
    __syncthreads();

    // --- scores: two 16x16 tiles (keys kt..kt+15, kt+16..kt+31)
    v8f s0 = vzero, s1 = vzero;
    s0 = wmma_bf16(q0, load_frag(sK, ln16, 64, 0), s0);
    s0 = wmma_bf16(q1, load_frag(sK, ln16, 64, 32), s0);
    s1 = wmma_bf16(q0, load_frag(sK, 16 + ln16, 64, 0), s1);
    s1 = wmma_bf16(q1, load_frag(sK, 16 + ln16, 64, 32), s1);

    // --- online softmax (rows live on accumulator VGPR index + lane half)
#pragma unroll
    for (int r = 0; r < 8; ++r) {
      const int qr = qchunk * 128 + w * 16 + r + rowAdd;
      const float* mp = mask + (((size_t)b * S_) + qr) * S_ + kt + ln16;
      float a0 = s0[r] * 0.125f + mp[0];
      float a1 = s1[r] * 0.125f + mp[16];
      float t = fmaxf(a0, a1);
#pragma unroll
      for (int sh = 1; sh < 16; sh <<= 1) t = fmaxf(t, __shfl_xor(t, sh, 32));
      const float mn = fmaxf(mrow[r], t);
      const float alpha = __expf(mrow[r] - mn);
      const float p0 = __expf(a0 - mn);
      const float p1 = __expf(a1 - mn);
      float rs = p0 + p1;
#pragma unroll
      for (int sh = 1; sh < 16; sh <<= 1) rs += __shfl_xor(rs, sh, 32);
      lrow[r] = lrow[r] * alpha + rs;
      mrow[r] = mn;
#pragma unroll
      for (int n = 0; n < 4; ++n) o[n][r] *= alpha;
      // C-layout -> A-layout via per-wave LDS buffer
      sP[w][(r + rowAdd) * 32 + ln16] = (bf16)p0;
      sP[w][(r + rowAdd) * 32 + 16 + ln16] = (bf16)p1;
    }

    // --- O += P(16x32) * V(32x64)
    const v16bf pa = load_frag(sP[w], ln16, 32, 0);
#pragma unroll
    for (int n = 0; n < 4; ++n)
      o[n] = wmma_bf16(pa, load_frag(sVt, n * 16 + ln16, 32, 0), o[n]);
  }

  // --- normalize and write bf16 output
#pragma unroll
  for (int n = 0; n < 4; ++n) {
#pragma unroll
    for (int r = 0; r < 8; ++r) {
      const int qr = qchunk * 128 + w * 16 + r + rowAdd;
      const float v = o[n][r] / lrow[r];
      vals[((size_t)(b * S_ + qr)) * D_ + h * HD_ + n * 16 + ln16] = (bf16)v;
    }
  }
}

// ---------------------------------------------------------------------------
// Residual add + LayerNorm. One block per token row; writes f32 + bf16.
// ---------------------------------------------------------------------------
__global__ __launch_bounds__(256) void add_ln_kernel(
    const float* __restrict__ x, const float* __restrict__ y,
    const float* __restrict__ gamma, const float* __restrict__ beta,
    float* __restrict__ xo, bf16* __restrict__ xob) {
  __shared__ float red[8];
  const int tid = threadIdx.x;
  const int lane = tid & 31;
  const int w = tid >> 5;
  const size_t base = (size_t)blockIdx.x * D_;

  float v[4];
  float sum = 0.0f;
#pragma unroll
  for (int j = 0; j < 4; ++j) {
    const int d = tid + j * 256;
    v[j] = x[base + d] + y[base + d];
    sum += v[j];
  }
#pragma unroll
  for (int sh = 1; sh < 32; sh <<= 1) sum += __shfl_xor(sum, sh, 32);
  if (lane == 0) red[w] = sum;
  __syncthreads();
  float tot = red[0];
#pragma unroll
  for (int i = 1; i < 8; ++i) tot += red[i];
  const float mean = tot * (1.0f / D_);

  float va = 0.0f;
#pragma unroll
  for (int j = 0; j < 4; ++j) { const float d2 = v[j] - mean; va += d2 * d2; }
#pragma unroll
  for (int sh = 1; sh < 32; sh <<= 1) va += __shfl_xor(va, sh, 32);
  __syncthreads();
  if (lane == 0) red[w] = va;
  __syncthreads();
  float vtot = red[0];
#pragma unroll
  for (int i = 1; i < 8; ++i) vtot += red[i];
  const float rstd = rsqrtf(vtot * (1.0f / D_) + 1e-5f);

#pragma unroll
  for (int j = 0; j < 4; ++j) {
    const int d = tid + j * 256;
    const float ov = gamma[d] * (v[j] - mean) * rstd + beta[d];
    xo[base + d] = ov;
    xob[base + d] = (bf16)ov;
  }
}

// ---------------------------------------------------------------------------
// Host-side orchestration
// ---------------------------------------------------------------------------
static inline void launch_gemm(hipStream_t s, const bf16* A, const bf16* Bt,
                               const float* bias, float* Cf, bf16* Cb,
                               int M, int N, int K, int relu) {
  dim3 grid(N / 256, M / 128);
  gemm_wmma_kernel<<<grid, dim3(256), 0, s>>>(A, Bt, bias, Cf, Cb, M, N, K, relu);
}

static inline void launch_transpose(hipStream_t s, const float* W, bf16* Wt,
                                    int K, int N) {
  size_t total = (size_t)K * N;
  transpose_w_kernel<<<dim3((unsigned)((total + 255) / 256)), dim3(256), 0, s>>>(
      W, Wt, K, N);
}

extern "C" void kernel_launch(void* const* d_in, const int* in_sizes, int n_in,
                              void* d_out, int out_size, void* d_ws, size_t ws_size,
                              hipStream_t stream) {
  (void)in_sizes; (void)n_in; (void)out_size; (void)ws_size;

  const int*   tokens = (const int*)d_in[0];
  const float* mask   = (const float*)d_in[1];
  const float* emb    = (const float*)d_in[2];
  const float* qkv_w  = (const float*)d_in[3];
  const float* qkv_b  = (const float*)d_in[4];
  const float* out_w  = (const float*)d_in[5];
  const float* out_b  = (const float*)d_in[6];
  const float* w1     = (const float*)d_in[7];
  const float* b1     = (const float*)d_in[8];
  const float* w2     = (const float*)d_in[9];
  const float* b2     = (const float*)d_in[10];
  const float* gamma  = (const float*)d_in[11];
  const float* beta   = (const float*)d_in[12];
  float* out = (float*)d_out;

  char* ws = (char*)d_ws;
  size_t off = 0;
  auto alloc = [&](size_t bytes) -> char* {
    char* p = ws + off;
    off = (off + bytes + 255) & ~(size_t)255;
    return p;
  };

  bf16*  xb    = (bf16*)alloc((size_t)BS_ * D_ * sizeof(bf16));
  float* x32   = (float*)alloc((size_t)BS_ * D_ * sizeof(float));
  bf16*  qkvb  = (bf16*)alloc((size_t)BS_ * 3 * D_ * sizeof(bf16));
  bf16*  valsb = (bf16*)alloc((size_t)BS_ * D_ * sizeof(bf16));
  float* y32   = (float*)alloc((size_t)BS_ * D_ * sizeof(float));
  bf16*  h1b   = (bf16*)alloc((size_t)BS_ * F_ * sizeof(bf16));
  bf16*  qkvT  = (bf16*)alloc((size_t)L_ * 3 * D_ * D_ * sizeof(bf16));
  bf16*  outT  = (bf16*)alloc((size_t)L_ * D_ * D_ * sizeof(bf16));
  bf16*  w1T   = (bf16*)alloc((size_t)L_ * F_ * D_ * sizeof(bf16));
  bf16*  w2T   = (bf16*)alloc((size_t)L_ * D_ * F_ * sizeof(bf16));

  // Pre-transpose + bf16-pack all weights (deterministic, every launch).
  for (int i = 0; i < L_; ++i) {
    launch_transpose(stream, qkv_w + (size_t)i * D_ * 3 * D_,
                     qkvT + (size_t)i * 3 * D_ * D_, D_, 3 * D_);
    launch_transpose(stream, out_w + (size_t)i * D_ * D_,
                     outT + (size_t)i * D_ * D_, D_, D_);
    launch_transpose(stream, w1 + (size_t)i * D_ * F_,
                     w1T + (size_t)i * F_ * D_, D_, F_);
    launch_transpose(stream, w2 + (size_t)i * F_ * D_,
                     w2T + (size_t)i * D_ * F_, F_, D_);
  }

  embed_pe_kernel<<<dim3(BS_), dim3(256), 0, stream>>>(tokens, emb, x32, xb);

  for (int i = 0; i < L_; ++i) {
    const float* g  = gamma + (size_t)i * D_;
    const float* bb = beta + (size_t)i * D_;

    // QKV projection -> bf16 [B,S,3D]
    launch_gemm(stream, xb, qkvT + (size_t)i * 3 * D_ * D_,
                qkv_b + (size_t)i * 3 * D_, nullptr, qkvb,
                BS_, 3 * D_, D_, 0);

    // Flash attention -> bf16 [B,S,D]
    attn_kernel<<<dim3(S_ / 128, B_ * H_), dim3(256), 0, stream>>>(
        qkvb, mask, valsb);

    // Output projection -> f32
    launch_gemm(stream, valsb, outT + (size_t)i * D_ * D_,
                out_b + (size_t)i * D_, y32, nullptr, BS_, D_, D_, 0);

    // x = LN(x + attn_out)
    add_ln_kernel<<<dim3(BS_), dim3(256), 0, stream>>>(x32, y32, g, bb, x32, xb);

    // FFN1 (+ReLU) -> bf16 [B,S,F]
    launch_gemm(stream, xb, w1T + (size_t)i * F_ * D_,
                b1 + (size_t)i * F_, nullptr, h1b, BS_, F_, D_, 1);

    // FFN2 -> f32
    launch_gemm(stream, h1b, w2T + (size_t)i * D_ * F_,
                b2 + (size_t)i * D_, y32, nullptr, BS_, D_, F_, 0);

    // x = LN(x + ff); final layer writes straight to d_out
    float* xo = (i == L_ - 1) ? out : x32;
    add_ln_kernel<<<dim3(BS_), dim3(256), 0, stream>>>(x32, y32, g, bb, xo, xb);
  }
}
